// self_transformer_41575283425956
// MI455X (gfx1250) — matrix operand order repaired and checked by reading, not compiled, not run on hardware
//
#include <hip/hip_runtime.h>
#include <cmath>

// ---------------------------------------------------------------------------
// CDNA5 (gfx1250, wave32) implementation of:
//   q = x@W1^T+b1; k = x@W2^T+b2; v = x@W3^T+b3
//   attn = softmax((k @ q^T) / sqrt(Dout)); out = attn @ v
// All GEMMs via v_wmma_f32_16x16x32_bf16 (fp32 accumulate).
// Wave tile 64x64 (16 WMMA accumulators), block tile 128x256 (8 waves).
// ---------------------------------------------------------------------------

typedef __attribute__((ext_vector_type(16))) __bf16 v16bf;
typedef __attribute__((ext_vector_type(8)))  float  v8f;

struct B32x2 { uint4 a, b; };

// Build a 16-element bf16 fragment from two 16-byte global chunks.
__device__ __forceinline__ v16bf load_frag(const __bf16* p0, const __bf16* p1) {
  B32x2 t;
  t.a = *(const uint4*)p0;
  t.b = *(const uint4*)p1;
  return __builtin_bit_cast(v16bf, t);
}

// ---------------------------------------------------------------------------
// fp32 -> bf16 conversion (n % 4 == 0)
// ---------------------------------------------------------------------------
__global__ void cvt_f32_bf16(const float* __restrict__ in, __bf16* __restrict__ out, int n) {
  int i = (blockIdx.x * blockDim.x + threadIdx.x) * 4;
  if (i < n) {
    float4 v = *(const float4*)(in + i);
    out[i + 0] = (__bf16)v.x;
    out[i + 1] = (__bf16)v.y;
    out[i + 2] = (__bf16)v.z;
    out[i + 3] = (__bf16)v.w;
  }
}

// ---------------------------------------------------------------------------
// Generic bf16 WMMA GEMM:  C[M, Ncols] = A[M,K] @ B[Ncols,K]^T
//   epilogue: v = acc*scale; if rowscale: v *= rowscale[row]; if bias: v += bias[col]
//   output either bf16 (outb) or fp32 (outf).
// Block: 256 threads (8 waves, 2x4). Block tile 128x256. Wave tile 64x64
// (4x4 WMMA tiles = 16 accumulators). Requires M%128==0, Ncols%256==0, K%32==0.
// ---------------------------------------------------------------------------
__global__ void __launch_bounds__(256)
gemm_bf16(const __bf16* __restrict__ A, const __bf16* __restrict__ B,
          const float* __restrict__ bias, const float* __restrict__ rowscale,
          float scale, __bf16* __restrict__ outb, float* __restrict__ outf,
          int M, int Ncols, int K) {
  const int lane  = threadIdx.x & 31;
  const int wave  = threadIdx.x >> 5;
  const int waveM = wave >> 2;         // 0..1
  const int waveN = wave & 3;          // 0..3
  const int l16   = lane & 15;
  const int lhalf = lane >> 4;         // 0 or 1

  const int wRow = blockIdx.y * 128 + waveM * 64;   // 64 rows per wave
  const int wCol = blockIdx.x * 256 + waveN * 64;   // 64 cols per wave

  // A 16x32 bf16 fragment layout (ISA 7.12.2):
  //  lane<16 : row = l16, K chunks [0..7] and [16..23]
  //  lane>=16: row = l16, K chunks [8..15] and [24..31]
  const __bf16* aRow[4];
#pragma unroll
  for (int mt = 0; mt < 4; ++mt)
    aRow[mt] = A + (size_t)(wRow + mt * 16 + l16) * K + lhalf * 8;

  // B 32x16 bf16 fragment layout: lane<16 holds K=[0..7],[16..23] of col=l16,
  // lane>=16 holds K=[8..15],[24..31] of col=l16 (same order as A).
  const __bf16* bRow[4];
#pragma unroll
  for (int nt = 0; nt < 4; ++nt)
    bRow[nt] = B + (size_t)(wCol + nt * 16 + l16) * K + lhalf * 8;

  v8f acc[4][4] = {};

  for (int k0 = 0; k0 < K; k0 += 32) {
    // Prefetch the k+64 chunks (CDNA5 global_prefetch_b8 path).
    if (k0 + 64 < K) {
      __builtin_prefetch(aRow[0] + k0 + 64, 0, 1);
      __builtin_prefetch(bRow[0] + k0 + 64, 0, 1);
    }

    v16bf af[4], bf[4];
#pragma unroll
    for (int mt = 0; mt < 4; ++mt)
      af[mt] = load_frag(aRow[mt] + k0, aRow[mt] + k0 + 16);
#pragma unroll
    for (int nt = 0; nt < 4; ++nt)
      bf[nt] = load_frag(bRow[nt] + k0, bRow[nt] + k0 + 16);

#pragma unroll
    for (int mt = 0; mt < 4; ++mt)
#pragma unroll
      for (int nt = 0; nt < 4; ++nt)
        acc[mt][nt] = __builtin_amdgcn_wmma_f32_16x16x32_bf16(
            /*neg_a=*/false, af[mt], /*neg_b=*/false, bf[nt],
            /*c_mod=*/(short)0, acc[mt][nt], /*reuse_a=*/false, /*reuse_b=*/false);
  }

  // C/D layout: VGPR r -> row = r (lanes 0-15) or r+8 (lanes 16-31); col = lane%16.
#pragma unroll
  for (int mt = 0; mt < 4; ++mt) {
#pragma unroll
    for (int nt = 0; nt < 4; ++nt) {
      const int col = wCol + nt * 16 + l16;
      const float bv = bias ? bias[col] : 0.0f;
#pragma unroll
      for (int r = 0; r < 8; ++r) {
        const int row = wRow + mt * 16 + lhalf * 8 + r;
        float v = acc[mt][nt][r] * scale;
        if (rowscale) v *= rowscale[row];
        v += bv;
        const size_t idx = (size_t)row * Ncols + col;
        if (outf) outf[idx] = v;
        else      outb[idx] = (__bf16)v;
      }
    }
  }
}

// ---------------------------------------------------------------------------
// bf16 transpose: out[c, r] = in[r, c]; 32x32 LDS tiles, rows%32==0, cols%32==0
// ---------------------------------------------------------------------------
__global__ void transpose_bf16(const __bf16* __restrict__ in, __bf16* __restrict__ out,
                               int rows, int cols) {
  __shared__ __bf16 tile[32][33];
  const int c0 = blockIdx.x * 32, r0 = blockIdx.y * 32;
  const int tx = threadIdx.x & 31, ty = threadIdx.x >> 5;   // 8 rows per step
#pragma unroll
  for (int i = 0; i < 32; i += 8)
    tile[ty + i][tx] = in[(size_t)(r0 + ty + i) * cols + c0 + tx];
  __syncthreads();
#pragma unroll
  for (int i = 0; i < 32; i += 8)
    out[(size_t)(c0 + ty + i) * rows + r0 + tx] = tile[tx][ty + i];
}

// ---------------------------------------------------------------------------
// Row softmax over S (fp32, already scaled): P = exp(S - rowmax) (bf16),
// invsum[row] = 1/sum(exp). One 256-thread block per row.
// ---------------------------------------------------------------------------
__global__ void softmax_rows(const float* __restrict__ S, __bf16* __restrict__ P,
                             float* __restrict__ invsum, int N) {
  const int row = blockIdx.x;
  const float* s = S + (size_t)row * N;
  __bf16* p = P + (size_t)row * N;
  __shared__ float red[256];

  float mx = -3.0e38f;
  for (int j = threadIdx.x; j < N; j += 256) mx = fmaxf(mx, s[j]);
  red[threadIdx.x] = mx;
  __syncthreads();
  for (int off = 128; off > 0; off >>= 1) {
    if ((int)threadIdx.x < off)
      red[threadIdx.x] = fmaxf(red[threadIdx.x], red[threadIdx.x + off]);
    __syncthreads();
  }
  mx = red[0];
  __syncthreads();

  float sum = 0.0f;
  for (int j = threadIdx.x; j < N; j += 256) {
    float e = __expf(s[j] - mx);
    sum += e;
    p[j] = (__bf16)e;
  }
  red[threadIdx.x] = sum;
  __syncthreads();
  for (int off = 128; off > 0; off >>= 1) {
    if ((int)threadIdx.x < off)
      red[threadIdx.x] += red[threadIdx.x + off];
    __syncthreads();
  }
  if (threadIdx.x == 0) invsum[row] = 1.0f / red[0];
}

// ---------------------------------------------------------------------------
// Host-side orchestration
// ---------------------------------------------------------------------------
extern "C" void kernel_launch(void* const* d_in, const int* in_sizes, int n_in,
                              void* d_out, int out_size, void* d_ws, size_t ws_size,
                              hipStream_t stream) {
  const float* x  = (const float*)d_in[0];
  const float* W1 = (const float*)d_in[1];
  const float* b1 = (const float*)d_in[2];
  const float* W2 = (const float*)d_in[3];
  const float* b2 = (const float*)d_in[4];
  const float* W3 = (const float*)d_in[5];
  const float* b3 = (const float*)d_in[6];

  const int Dout = in_sizes[2];                 // 1024
  const int Din  = in_sizes[1] / Dout;          // 1024
  const int N    = in_sizes[0] / Din;           // 4096
  const float scale = 1.0f / sqrtf((float)Dout);

  // Workspace carve (256B aligned)
  char* ws = (char*)d_ws;
  size_t off = 0;
  auto carve = [&](size_t bytes) -> char* {
    char* p = ws + off;
    off += (bytes + 255) & ~(size_t)255;
    return p;
  };
  __bf16* Xb  = (__bf16*)carve((size_t)N * Din * 2);
  __bf16* W1b = (__bf16*)carve((size_t)Dout * Din * 2);
  __bf16* W2b = (__bf16*)carve((size_t)Dout * Din * 2);
  __bf16* W3b = (__bf16*)carve((size_t)Dout * Din * 2);
  __bf16* Qb  = (__bf16*)carve((size_t)N * Dout * 2);
  __bf16* Kb  = (__bf16*)carve((size_t)N * Dout * 2);
  __bf16* Vb  = (__bf16*)carve((size_t)N * Dout * 2);
  __bf16* Vt  = (__bf16*)carve((size_t)Dout * N * 2);
  float*  S   = (float*) carve((size_t)N * N * 4);
  __bf16* Pb  = (__bf16*)carve((size_t)N * N * 2);
  float*  inv = (float*) carve((size_t)N * 4);

  // 1) fp32 -> bf16 for X and weights
  {
    int n = N * Din;
    cvt_f32_bf16<<<(n / 4 + 255) / 256, 256, 0, stream>>>(x, Xb, n);
    n = Dout * Din;
    cvt_f32_bf16<<<(n / 4 + 255) / 256, 256, 0, stream>>>(W1, W1b, n);
    cvt_f32_bf16<<<(n / 4 + 255) / 256, 256, 0, stream>>>(W2, W2b, n);
    cvt_f32_bf16<<<(n / 4 + 255) / 256, 256, 0, stream>>>(W3, W3b, n);
  }

  // 2) Projections: Q/K/V [N, Dout] = Xb @ Wb^T + b  (bf16 out)
  {
    dim3 grid(Dout / 256, N / 128);
    gemm_bf16<<<grid, 256, 0, stream>>>(Xb, W1b, b1, nullptr, 1.0f, Qb, nullptr, N, Dout, Din);
    gemm_bf16<<<grid, 256, 0, stream>>>(Xb, W2b, b2, nullptr, 1.0f, Kb, nullptr, N, Dout, Din);
    gemm_bf16<<<grid, 256, 0, stream>>>(Xb, W3b, b3, nullptr, 1.0f, Vb, nullptr, N, Dout, Din);
  }

  // 3) Vt[Dout, N] = V^T (so attn@V B-fragments are contiguous over K=m)
  {
    dim3 grid(Dout / 32, N / 32);
    transpose_bf16<<<grid, 256, 0, stream>>>(Vb, Vt, N, Dout);
  }

  // 4) Scores: S[N, N] = scale * (Kb @ Qb^T)   (fp32 out)
  {
    dim3 grid(N / 256, N / 128);
    gemm_bf16<<<grid, 256, 0, stream>>>(Kb, Qb, nullptr, nullptr, scale, nullptr, S, N, N, Din);
  }

  // 5) Row softmax: P = exp(S - rowmax) bf16; inv = 1/rowsum
  softmax_rows<<<N, 256, 0, stream>>>(S, Pb, inv, N);

  // 6) out[N, Dout] = (P @ Vt^T) * inv[row]   (fp32 to d_out)
  {
    dim3 grid(Dout / 256, N / 128);
    gemm_bf16<<<grid, 256, 0, stream>>>(Pb, Vt, nullptr, inv, 1.0f, nullptr, (float*)d_out,
                                        N, Dout, N);
  }

  (void)n_in; (void)out_size; (void)ws_size;
}
